// ManyToManyRecurrentBase_10041633538900
// MI455X (gfx1250) — matrix operand order repaired and checked by reading, not compiled, not run on hardware
//
#include <hip/hip_runtime.h>
#include <hip/hip_bf16.h>

// Problem constants (match reference).
#define Bdim 256
#define Sdim 1024
#define Idim 128
#define Hdim 1024
#define Odim 64
#define Fdim 8

#define KBLK 64          // K elements staged per LDS block
#define NBH  (Hdim / KBLK)   // 16 hidden-phase K-blocks
#define NBX  (Idim / KBLK)   // 2 input-phase K-blocks

typedef __attribute__((ext_vector_type(16))) __bf16 bf16x16;
typedef __attribute__((ext_vector_type(8)))  __bf16 bf16x8;
typedef __attribute__((ext_vector_type(8)))  float  f32x8;

#if __has_builtin(__builtin_amdgcn_global_load_async_to_lds_b128) && \
    __has_builtin(__builtin_amdgcn_s_wait_asynccnt)
#define ASYNC_LDS_OK 1
#endif

// Exact parameter types of the async-copy builtin (from hipcc diagnostic):
//   (v4i addrspace(1)* src, v4i addrspace(3)* dst, imm offset, imm cpol)
typedef int v4i_t __attribute__((vector_size(16)));
typedef __attribute__((address_space(1))) v4i_t* gas_v4i;
typedef __attribute__((address_space(3))) v4i_t* las_v4i;

// 16-byte global -> LDS copy. Async (ASYNCcnt-tracked DMA, no VGPR round trip)
// when the CDNA5 builtin is available; synchronous load+ds_store otherwise.
static __device__ __forceinline__ void cp_g2l_16B(const __bf16* g, __bf16* l) {
#ifdef ASYNC_LDS_OK
  __builtin_amdgcn_global_load_async_to_lds_b128((gas_v4i)(void*)g,
                                                 (las_v4i)(void*)l, 0, 0);
#else
  *reinterpret_cast<bf16x8*>(l) = *reinterpret_cast<const bf16x8*>(g);
#endif
}

// ---------------------------------------------------------------------------
// Stage one K-block into LDS.
//   sA: [64 rows][KBLK]          A tile (hidden state or x_t rows)
//   sB: [3*64 rows][KBLK]        B tiles for gates r,z,n (rows of W)
// 256 threads; 8 x 16B chunks per thread (2 for A, 6 for B).
// ---------------------------------------------------------------------------
static __device__ __forceinline__ void
stage_block(__bf16* sA, __bf16* sB,
            const __bf16* __restrict__ gA, size_t ldA, int m0, int k0,
            const __bf16* __restrict__ gW, size_t ldW, int j0, int tid) {
#pragma unroll
  for (int i = 0; i < 2; ++i) {                 // A: 512 chunks
    const int c = tid + i * 256;
    const int row = c >> 3, kc = (c & 7) * 8;
    cp_g2l_16B(gA + (size_t)(m0 + row) * ldA + k0 + kc, sA + row * KBLK + kc);
  }
#pragma unroll
  for (int i = 0; i < 6; ++i) {                 // B: 1536 chunks
    const int c = tid + i * 256;
    const int g = c >> 9, n = (c >> 3) & 63, kc = (c & 7) * 8;
    cp_g2l_16B(gW + (size_t)(g * Hdim + j0 + n) * ldW + k0 + kc,
               sB + (g * 64 + n) * KBLK + kc);
  }
}

// ---------------------------------------------------------------------------
// Fragment loaders per CDNA5 WMMA VGPR layouts (CDNA5 ISA §7.12.2)
// A (16x32 bf16): lane L<16 -> row L, elems 0-7 = K 0-7, elems 8-15 = K 16-23;
//                 lane L>=16 -> row L-16, K offsets +8.
// B (32x16 bf16): lane L<16 -> col L, elems 0-7 = K 0-7, elems 8-15 = K 16-23; lane L>=16 -> col L-16, K offsets +8.
// ---------------------------------------------------------------------------
static __device__ __forceinline__ bf16x16
lds_a_frag(const __bf16* sA, int row0, int ko, int lane) {
  const int r  = row0 + (lane & 15);
  const int kb = ko + ((lane & 16) ? 8 : 0);
  const __bf16* p = sA + r * KBLK + kb;
  bf16x8 lo = *reinterpret_cast<const bf16x8*>(p);
  bf16x8 hi = *reinterpret_cast<const bf16x8*>(p + 16);
  return __builtin_shufflevector(lo, hi, 0,1,2,3,4,5,6,7,8,9,10,11,12,13,14,15);
}

static __device__ __forceinline__ bf16x16
lds_b_frag(const __bf16* sB, int gate, int n0, int ko, int lane) {
  const int n  = n0 + (lane & 15);
  const int kb = ko + ((lane & 16) ? 8 : 0);
  const __bf16* p = sB + (gate * 64 + n) * KBLK + kb;
  bf16x8 lo = *reinterpret_cast<const bf16x8*>(p);
  bf16x8 hi = *reinterpret_cast<const bf16x8*>(p + 16);
  return __builtin_shufflevector(lo, hi, 0,1,2,3,4,5,6,7,8,9,10,11,12,13,14,15);
}

static __device__ __forceinline__ float sigmoid_fast(float x) {
  return 1.0f / (1.0f + __expf(-x));
}

#define WMMA_BF16(A, Bv, C) \
  __builtin_amdgcn_wmma_f32_16x16x32_bf16(false, (A), false, (Bv), (short)0, (C), false, false)

// ---------------------------------------------------------------------------
// One fused GRU step: gates = h @ W_hh^T (+ x_t @ W_ih^T) + biases,
//                     h' = (1-z)*n + z*h.
// Grid (Bdim/64, Hdim/64) = (4,16); block 256 = 8 waves (2 M-waves x 4 N-waves).
// Each wave owns a 32x16 tile (2 M-positions sharing every B fragment).
// LDS double buffer: next K-block streamed with async global->LDS DMA while
// the current block feeds 12 WMMAs per wave. xb_t == nullptr -> future step.
// ---------------------------------------------------------------------------
__global__ void __launch_bounds__(256)
gru_step_wmma(const __hip_bfloat16* __restrict__ hb_in_,
              const float*          __restrict__ hf_in,
              const __hip_bfloat16* __restrict__ xb_t_,
              const __hip_bfloat16* __restrict__ whh_,
              const __hip_bfloat16* __restrict__ wih_,
              const float*          __restrict__ b_ih,
              const float*          __restrict__ b_hh,
              float*                __restrict__ hf_out,
              __hip_bfloat16*       __restrict__ hb_out) {
  __shared__ __bf16 sA[2][64 * KBLK];       // 2 x 8 KB
  __shared__ __bf16 sB[2][3 * 64 * KBLK];   // 2 x 24 KB  (64 KB total LDS)

  const __bf16* hb  = reinterpret_cast<const __bf16*>(hb_in_);
  const __bf16* xbt = reinterpret_cast<const __bf16*>(xb_t_);
  const __bf16* whh = reinterpret_cast<const __bf16*>(whh_);
  const __bf16* wih = reinterpret_cast<const __bf16*>(wih_);

  const int tid  = threadIdx.x;
  const int lane = tid & 31;
  const int wave = tid >> 5;
  const int wm = wave >> 2;                 // 0..1
  const int wn = wave & 3;                  // 0..3
  const int m0wg = blockIdx.x * 64;         // batch-row base
  const int j0wg = blockIdx.y * 64;         // hidden-col base

  // C/D layout: lane -> N = lane&15 (same for all 8 acc VGPRs) -> per-lane biases.
  const int   j   = j0wg + wn * 16 + (lane & 15);
  const float brv = b_ih[j]            + b_hh[j];
  const float bzv = b_ih[Hdim + j]     + b_hh[Hdim + j];
  const float bxn = b_ih[2 * Hdim + j];     // input part of n-gate
  const float bhn = b_hh[2 * Hdim + j];     // hidden part of n-gate (scaled by r)

  // acc[pos][0]=r, [1]=z, [2]=xn, [3]=hn
  f32x8 acc[2][4];
#pragma unroll
  for (int p = 0; p < 2; ++p)
#pragma unroll
    for (int v = 0; v < 8; ++v) {
      acc[p][0][v] = brv; acc[p][1][v] = bzv; acc[p][2][v] = bxn; acc[p][3][v] = bhn;
    }

  const int NB = (xbt != nullptr) ? (NBH + NBX) : NBH;   // 18 observed / 16 future

  // Prime stage 0 (hidden phase always exists).
  stage_block(&sA[0][0], &sB[0][0], hb, Hdim, m0wg, 0, whh, Hdim, j0wg, tid);

  for (int kb = 0; kb < NB; ++kb) {
    const int s = kb & 1;
    if (kb + 1 < NB) {                      // prefetch next K-block into other buffer
      const int kn = kb + 1;
      if (kn < NBH)
        stage_block(&sA[s ^ 1][0], &sB[s ^ 1][0], hb, Hdim, m0wg, kn * KBLK,
                    whh, Hdim, j0wg, tid);
      else
        stage_block(&sA[s ^ 1][0], &sB[s ^ 1][0], xbt, (size_t)Sdim * Idim, m0wg,
                    (kn - NBH) * KBLK, wih, Idim, j0wg, tid);
#ifdef ASYNC_LDS_OK
      __builtin_amdgcn_s_wait_asynccnt(8);  // stage kb done; stage kb+1 (8 ops) in flight
#endif
    } else {
#ifdef ASYNC_LDS_OK
      __builtin_amdgcn_s_wait_asynccnt(0);
#endif
    }
    __syncthreads();                        // stage kb visible to all waves

    const bool xph = (kb >= NBH);           // uniform: EXEC stays all-ones
#pragma unroll
    for (int kk = 0; kk < KBLK / 32; ++kk) {
      const int ko = kk * 32;
      bf16x16 a0 = lds_a_frag(&sA[s][0], wm * 32,      ko, lane);
      bf16x16 a1 = lds_a_frag(&sA[s][0], wm * 32 + 16, ko, lane);
      bf16x16 bR = lds_b_frag(&sB[s][0], 0, wn * 16, ko, lane);
      bf16x16 bZ = lds_b_frag(&sB[s][0], 1, wn * 16, ko, lane);
      bf16x16 bN = lds_b_frag(&sB[s][0], 2, wn * 16, ko, lane);
      acc[0][0] = WMMA_BF16(a0, bR, acc[0][0]);
      acc[1][0] = WMMA_BF16(a1, bR, acc[1][0]);
      acc[0][1] = WMMA_BF16(a0, bZ, acc[0][1]);
      acc[1][1] = WMMA_BF16(a1, bZ, acc[1][1]);
      if (!xph) {
        acc[0][3] = WMMA_BF16(a0, bN, acc[0][3]);
        acc[1][3] = WMMA_BF16(a1, bN, acc[1][3]);
      } else {
        acc[0][2] = WMMA_BF16(a0, bN, acc[0][2]);
        acc[1][2] = WMMA_BF16(a1, bN, acc[1][2]);
      }
    }
    __syncthreads();                        // all reads done before buffer reuse
  }

  // Gate epilogue (fp32). C layout: lane -> N, VGPR v -> M = v + (lane>=16 ? 8:0).
#pragma unroll
  for (int p = 0; p < 2; ++p) {
    const int mb = m0wg + wm * 32 + p * 16 + ((lane & 16) ? 8 : 0);
#pragma unroll
    for (int v = 0; v < 8; ++v) {
      const int   m    = mb + v;
      const float hold = hf_in[(size_t)m * Hdim + j];
      const float r    = sigmoid_fast(acc[p][0][v]);
      const float z    = sigmoid_fast(acc[p][1][v]);
      const float n    = tanhf(acc[p][2][v] + r * acc[p][3][v]);
      const float hnew = n + z * (hold - n);          // (1-z)*n + z*h
      hf_out[(size_t)m * Hdim + j] = hnew;
      hb_out[(size_t)m * Hdim + j] = __float2bfloat16(hnew);
    }
  }
}

// ---------------------------------------------------------------------------
// Helpers: fp32 -> bf16 conversion, state zero-init, fp32 output projection.
// ---------------------------------------------------------------------------
__global__ void cvt_f32_bf16(const float* __restrict__ s, __hip_bfloat16* __restrict__ d, int n) {
  const int i = blockIdx.x * blockDim.x + threadIdx.x;
  if (i < n) d[i] = __float2bfloat16(s[i]);
}

__global__ void zero_state(float* __restrict__ hf, __hip_bfloat16* __restrict__ hbv, int n) {
  const int i = blockIdx.x * blockDim.x + threadIdx.x;
  if (i < n) { hf[i] = 0.0f; hbv[i] = __float2bfloat16(0.0f); }
}

// out[b, f, :] = h[b,:] @ W_out^T + b_out   (tiny: 256x64 dots of length 1024, fp32)
__global__ void out_proj(const float* __restrict__ hf, const float* __restrict__ W_out,
                         const float* __restrict__ b_out, float* __restrict__ out, int f) {
  const int b = blockIdx.x;
  const int o = threadIdx.x;
  const float4* hrow = reinterpret_cast<const float4*>(hf + (size_t)b * Hdim);
  const float4* wrow = reinterpret_cast<const float4*>(W_out + (size_t)o * Hdim);
  float acc = b_out[o];
#pragma unroll 4
  for (int k = 0; k < Hdim / 4; ++k) {
    const float4 hv = hrow[k];
    const float4 wv = wrow[k];
    acc += hv.x * wv.x + hv.y * wv.y + hv.z * wv.z + hv.w * wv.w;
  }
  out[((size_t)b * Fdim + f) * Odim + o] = acc;
}

// ---------------------------------------------------------------------------
extern "C" void kernel_launch(void* const* d_in, const int* in_sizes, int n_in,
                              void* d_out, int out_size, void* d_ws, size_t ws_size,
                              hipStream_t stream) {
  (void)in_sizes; (void)n_in; (void)out_size; (void)ws_size;
  const float* x     = (const float*)d_in[0];
  const float* W_ih  = (const float*)d_in[1];
  const float* W_hh  = (const float*)d_in[2];
  const float* b_ih  = (const float*)d_in[3];
  const float* b_hh  = (const float*)d_in[4];
  const float* W_out = (const float*)d_in[5];
  const float* b_out = (const float*)d_in[6];
  // d_in[7] = future_steps (device scalar); compile-time Fdim matches reference.
  float* out = (float*)d_out;

  // Workspace carve (all chunks 256B aligned): ~73.8 MB total.
  char* w = (char*)d_ws;
  __hip_bfloat16* xb   = (__hip_bfloat16*)w; w += (size_t)Bdim * Sdim * Idim * 2; // 64 MB
  __hip_bfloat16* whhb = (__hip_bfloat16*)w; w += (size_t)3 * Hdim * Hdim * 2;    // 6 MB
  __hip_bfloat16* wihb = (__hip_bfloat16*)w; w += (size_t)3 * Hdim * Idim * 2;    // 768 KB
  float*          hf0  = (float*)w;          w += (size_t)Bdim * Hdim * 4;        // 1 MB
  float*          hf1  = (float*)w;          w += (size_t)Bdim * Hdim * 4;        // 1 MB
  __hip_bfloat16* hb0  = (__hip_bfloat16*)w; w += (size_t)Bdim * Hdim * 2;        // 512 KB
  __hip_bfloat16* hb1  = (__hip_bfloat16*)w; w += (size_t)Bdim * Hdim * 2;        // 512 KB

  // Prep: bf16 conversions + zero initial state (re-done every call: deterministic).
  {
    int n = Bdim * Sdim * Idim;
    cvt_f32_bf16<<<(n + 255) / 256, 256, 0, stream>>>(x, xb, n);
    n = 3 * Hdim * Hdim;
    cvt_f32_bf16<<<(n + 255) / 256, 256, 0, stream>>>(W_hh, whhb, n);
    n = 3 * Hdim * Idim;
    cvt_f32_bf16<<<(n + 255) / 256, 256, 0, stream>>>(W_ih, wihb, n);
    n = Bdim * Hdim;
    zero_state<<<(n + 255) / 256, 256, 0, stream>>>(hf0, hb0, n);
  }

  const dim3 grid(Bdim / 64, Hdim / 64);   // 4 x 16 = 64 workgroups
  const dim3 block(256);

  float* hf_cur = hf0;          float* hf_nxt = hf1;
  __hip_bfloat16* hb_cur = hb0; __hip_bfloat16* hb_nxt = hb1;

  // Observed sequence: 1024 fused GEMM+gate steps (captured into the graph).
  for (int t = 0; t < Sdim; ++t) {
    gru_step_wmma<<<grid, block, 0, stream>>>(hb_cur, hf_cur, xb + (size_t)t * Idim,
                                              whhb, wihb, b_ih, b_hh, hf_nxt, hb_nxt);
    float* tf = hf_cur; hf_cur = hf_nxt; hf_nxt = tf;
    __hip_bfloat16* tb = hb_cur; hb_cur = hb_nxt; hb_nxt = tb;
  }
  out_proj<<<dim3(Bdim), dim3(Odim), 0, stream>>>(hf_cur, W_out, b_out, out, 0);

  // Future steps: dummy input -> skip input phase (biases b_ih already folded in).
  for (int f = 1; f < Fdim; ++f) {
    gru_step_wmma<<<grid, block, 0, stream>>>(hb_cur, hf_cur, nullptr,
                                              whhb, wihb, b_ih, b_hh, hf_nxt, hb_nxt);
    float* tf = hf_cur; hf_cur = hf_nxt; hf_nxt = tf;
    __hip_bfloat16* tb = hb_cur; hb_cur = hb_nxt; hb_nxt = tb;
    out_proj<<<dim3(Bdim), dim3(Odim), 0, stream>>>(hf_cur, W_out, b_out, out, f);
  }
}
